// DreamGenerator_14508399526507
// MI455X (gfx1250) — matrix operand order repaired and checked by reading, not compiled, not run on hardware
//
#include <hip/hip_runtime.h>
#include <hip/hip_bf16.h>

// ---------------- constants ----------------
#define CD   1024         // D
#define CE   4            // experts
#define CL   2            // layers
#define CFF  4096         // FF
#define CB   512          // batch
#define CND  3            // num dreams
#define CBT  (CND * CB)   // 1536
#define CST  5            // STEPS
#define CH   8            // heads
#define CHD  128          // head dim

typedef __attribute__((ext_vector_type(16))) __bf16 v16bf;
typedef __attribute__((ext_vector_type(8)))  float  v8f;

__device__ __forceinline__ float gelu_exact(float x) {
    return 0.5f * x * (1.0f + erff(x * 0.70710678118654752440f));
}

// ---------------- WMMA GEMM: Out = act(A[M,K] @ W[N,K]^T + bias) (+Res) ----------------
// REQUIRES: M % 64 == 0, N % 128 == 0, K % 32 == 0 (true for every launch below).
// block = 128 threads (4 waves). Block tile 64x128, wave tile 32x64 (8 WMMAs/K-step).
// Double-buffered LDS ping-pong: next tile's global loads issue before the WMMAs,
// convert+store lands in the other buffer, one barrier per K-step.
#define BMT 64
#define BNT 128
#define LDSS 36   // bf16 row stride (32 + pad, 4B-aligned rows, conflict-free frag reads)

__global__ __launch_bounds__(128)
void gemm_bf16_wmma(const float* __restrict__ A, const float* __restrict__ W,
                    const float* __restrict__ bias, const float* __restrict__ Res,
                    float* __restrict__ Out, int M, int N, int K, int act)
{
    __shared__ __bf16 As[2][BMT * LDSS];
    __shared__ __bf16 Bs[2][BNT * LDSS];

    const int tid  = threadIdx.x;
    const int wave = tid >> 5;
    const int lane = tid & 31;
    const int half = lane >> 4;
    const int l15  = lane & 15;

    const int m0 = blockIdx.y * BMT;
    const int n0 = blockIdx.x * BNT;
    const int mw = (wave >> 1) * 32;   // wave row offset within block tile
    const int nw = (wave & 1) * 64;    // wave col offset within block tile

    v8f acc[2][4];
    #pragma unroll
    for (int r = 0; r < 2; ++r)
        #pragma unroll
        for (int t = 0; t < 4; ++t) acc[r][t] = (v8f){};

    // per-thread staging coords: chunk c -> row = c>>3, col4 = (c&7)*4
    float4 pa[4], pw[8];

    auto issue_loads = [&](int k0) {
        #pragma unroll
        for (int i = 0; i < 4; ++i) {
            int c = tid + 128 * i;
            pa[i] = *(const float4*)(A + (size_t)(m0 + (c >> 3)) * K + k0 + (c & 7) * 4);
        }
        #pragma unroll
        for (int i = 0; i < 8; ++i) {
            int c = tid + 128 * i;
            pw[i] = *(const float4*)(W + (size_t)(n0 + (c >> 3)) * K + k0 + (c & 7) * 4);
        }
    };
    auto store_lds = [&](int buf) {
        #pragma unroll
        for (int i = 0; i < 4; ++i) {
            int c = tid + 128 * i;
            __bf16* p = &As[buf][(c >> 3) * LDSS + (c & 7) * 4];
            p[0] = (__bf16)pa[i].x; p[1] = (__bf16)pa[i].y;
            p[2] = (__bf16)pa[i].z; p[3] = (__bf16)pa[i].w;
        }
        #pragma unroll
        for (int i = 0; i < 8; ++i) {
            int c = tid + 128 * i;
            __bf16* p = &Bs[buf][(c >> 3) * LDSS + (c & 7) * 4];
            p[0] = (__bf16)pw[i].x; p[1] = (__bf16)pw[i].y;
            p[2] = (__bf16)pw[i].z; p[3] = (__bf16)pw[i].w;
        }
    };

    issue_loads(0);
    store_lds(0);
    __syncthreads();

    const int nk = K >> 5;
    for (int ki = 0; ki < nk; ++ki) {
        const int buf = ki & 1;
        if (ki + 1 < nk) issue_loads((ki + 1) << 5);   // prefetch next tile (no wait yet)

        // fragments per CDNA5 16-bit layouts:
        // A 16x32: ka = half*8 + (j<8 ? j : j+8) ; B 32x16: kb = half*8 + (j<8 ? j : j+8)
        v16bf a0, a1, b0, b1, b2, b3;
        #pragma unroll
        for (int j = 0; j < 16; ++j) {
            int ka = half * 8 + (j < 8 ? j : j + 8);
            int kb = half * 8 + (j < 8 ? j : j + 8);
            a0[j] = As[buf][(mw + l15)      * LDSS + ka];
            a1[j] = As[buf][(mw + 16 + l15) * LDSS + ka];
            b0[j] = Bs[buf][(nw + l15)      * LDSS + kb];
            b1[j] = Bs[buf][(nw + 16 + l15) * LDSS + kb];
            b2[j] = Bs[buf][(nw + 32 + l15) * LDSS + kb];
            b3[j] = Bs[buf][(nw + 48 + l15) * LDSS + kb];
        }
        acc[0][0] = __builtin_amdgcn_wmma_f32_16x16x32_bf16(false, a0, false, b0, (short)0, acc[0][0], false, false);
        acc[0][1] = __builtin_amdgcn_wmma_f32_16x16x32_bf16(false, a0, false, b1, (short)0, acc[0][1], false, false);
        acc[0][2] = __builtin_amdgcn_wmma_f32_16x16x32_bf16(false, a0, false, b2, (short)0, acc[0][2], false, false);
        acc[0][3] = __builtin_amdgcn_wmma_f32_16x16x32_bf16(false, a0, false, b3, (short)0, acc[0][3], false, false);
        acc[1][0] = __builtin_amdgcn_wmma_f32_16x16x32_bf16(false, a1, false, b0, (short)0, acc[1][0], false, false);
        acc[1][1] = __builtin_amdgcn_wmma_f32_16x16x32_bf16(false, a1, false, b1, (short)0, acc[1][1], false, false);
        acc[1][2] = __builtin_amdgcn_wmma_f32_16x16x32_bf16(false, a1, false, b2, (short)0, acc[1][2], false, false);
        acc[1][3] = __builtin_amdgcn_wmma_f32_16x16x32_bf16(false, a1, false, b3, (short)0, acc[1][3], false, false);

        if (ki + 1 < nk) store_lds(buf ^ 1);           // waits on the prefetch, fills other buffer
        __syncthreads();
    }

    // epilogue: C/D f32 layout: VGPR r, lane -> row = r + 8*half, col = lane&15
    #pragma unroll
    for (int rt = 0; rt < 2; ++rt) {
        #pragma unroll
        for (int ct = 0; ct < 4; ++ct) {
            int col = n0 + nw + ct * 16 + l15;
            if (col >= N) continue;
            float bv = bias ? bias[col] : 0.0f;
            #pragma unroll
            for (int r = 0; r < 8; ++r) {
                int row = m0 + mw + rt * 16 + half * 8 + r;
                if (row >= M) continue;
                float v = acc[rt][ct][r] + bv;
                if (act) v = gelu_exact(v);
                size_t o = (size_t)row * N + col;
                if (Res) v += Res[o];
                Out[o] = v;
            }
        }
    }
}

// ---------------- LayerNorm over width 1024 (optionally fused GELU) ----------------
__global__ __launch_bounds__(256)
void k_ln(const float* __restrict__ x, float* __restrict__ y,
          const float* __restrict__ w, const float* __restrict__ b,
          long in_stride, long in_off, int do_gelu)
{
    __shared__ float rs[256], rs2[256];
    int row = blockIdx.x;
    int tid = threadIdx.x;
    const float* xr = x + (size_t)row * in_stride + in_off;
    float4 v = *(const float4*)(xr + tid * 4);
    rs[tid]  = v.x + v.y + v.z + v.w;
    rs2[tid] = v.x * v.x + v.y * v.y + v.z * v.z + v.w * v.w;
    __syncthreads();
    for (int st = 128; st > 0; st >>= 1) {
        if (tid < st) { rs[tid] += rs[tid + st]; rs2[tid] += rs2[tid + st]; }
        __syncthreads();
    }
    float mean = rs[0] * (1.0f / 1024.0f);
    float var  = rs2[0] * (1.0f / 1024.0f) - mean * mean;
    float inv  = rsqrtf(var + 1e-5f);
    int di = tid * 4;
    float4 o;
    o.x = (v.x - mean) * inv * w[di + 0] + b[di + 0];
    o.y = (v.y - mean) * inv * w[di + 1] + b[di + 1];
    o.z = (v.z - mean) * inv * w[di + 2] + b[di + 2];
    o.w = (v.w - mean) * inv * w[di + 3] + b[di + 3];
    if (do_gelu) {
        o.x = gelu_exact(o.x); o.y = gelu_exact(o.y);
        o.z = gelu_exact(o.z); o.w = gelu_exact(o.w);
    }
    *(float4*)(y + (size_t)row * 1024 + di) = o;
}

// ---------------- tiny MHA (t <= 5), one block per (batch-row, head) ----------------
__global__ __launch_bounds__(128)
void k_attn(const float* __restrict__ qkv, float* __restrict__ attn, int t)
{
    int bh = blockIdx.x;
    int b  = bh / CH;
    int h  = bh % CH;
    __shared__ float sq[CST][CHD], sk[CST][CHD], sv[CST][CHD], sp[CST][CST];
    int tid = threadIdx.x;               // 0..127 == head-dim element
    const float* base = qkv + (size_t)b * t * 3072 + h * CHD;
    for (int i = 0; i < t; ++i) {
        const float* rp = base + (size_t)i * 3072;
        sq[i][tid] = rp[tid];
        sk[i][tid] = rp[1024 + tid];
        sv[i][tid] = rp[2048 + tid];
    }
    __syncthreads();
    int lane = tid & 31, w = tid >> 5;   // 4 waves, each covers score pairs strided by 4
    for (int p = w; p < t * t; p += 4) {
        int i = p / t, j = p % t;
        float s = 0.f;
        for (int d = lane; d < CHD; d += 32) s += sq[i][d] * sk[j][d];
        s += __shfl_xor(s, 16); s += __shfl_xor(s, 8); s += __shfl_xor(s, 4);
        s += __shfl_xor(s, 2);  s += __shfl_xor(s, 1);
        if (lane == 0) sp[i][j] = s * 0.08838834764831845f;  // 1/sqrt(128)
    }
    __syncthreads();
    if (tid < t) {
        int i = tid;
        float mx = -1e30f;
        for (int j = 0; j < t; ++j) mx = fmaxf(mx, sp[i][j]);
        float sum = 0.f;
        for (int j = 0; j < t; ++j) { float ev = __expf(sp[i][j] - mx); sp[i][j] = ev; sum += ev; }
        float is = 1.0f / sum;
        for (int j = 0; j < t; ++j) sp[i][j] *= is;
    }
    __syncthreads();
    for (int i = 0; i < t; ++i) {
        float o = 0.f;
        for (int j = 0; j < t; ++j) o += sp[i][j] * sv[j][tid];
        attn[((size_t)b * t + i) * 1024 + h * CHD + tid] = o;
    }
}

// ---------------- gating: logits, top-2 softmax ----------------
__global__ __launch_bounds__(256)
void k_logits(const float* __restrict__ g, const float* __restrict__ gw2,
              const float* __restrict__ gb2, float* __restrict__ logits)
{
    __shared__ float red[4][256];
    int bi = blockIdx.x, tid = threadIdx.x;
    const float* gr = g + (size_t)bi * 1024;
    float p0 = 0, p1 = 0, p2 = 0, p3 = 0;
    for (int d = tid; d < 1024; d += 256) {
        float x = gr[d];
        p0 += x * gw2[d];        p1 += x * gw2[1024 + d];
        p2 += x * gw2[2048 + d]; p3 += x * gw2[3072 + d];
    }
    red[0][tid] = p0; red[1][tid] = p1; red[2][tid] = p2; red[3][tid] = p3;
    __syncthreads();
    for (int s = 128; s > 0; s >>= 1) {
        if (tid < s) { red[0][tid] += red[0][tid + s]; red[1][tid] += red[1][tid + s];
                       red[2][tid] += red[2][tid + s]; red[3][tid] += red[3][tid + s]; }
        __syncthreads();
    }
    if (tid < 4) logits[bi * 4 + tid] = red[tid][0] + gb2[tid];
}

__global__ void k_top2(const float* __restrict__ logits, float* __restrict__ wts)
{
    int b = blockIdx.x * blockDim.x + threadIdx.x;
    if (b >= CB) return;
    float l[4];
    for (int i = 0; i < 4; ++i) l[i] = logits[b * 4 + i];
    int i1 = 0;
    for (int i = 1; i < 4; ++i) if (l[i] > l[i1]) i1 = i;
    int i2 = -1;
    for (int i = 0; i < 4; ++i) { if (i == i1) continue; if (i2 < 0 || l[i] > l[i2]) i2 = i; }
    float m  = l[i1];
    float e1 = __expf(l[i1] - m), e2 = __expf(l[i2] - m);
    float s  = e1 + e2;
    float wv[4] = {0.f, 0.f, 0.f, 0.f};
    wv[i1] = e1 / s; wv[i2] = e2 / s;
    for (int i = 0; i < 4; ++i) wts[b * 4 + i] = wv[i];
}

// ---------------- elementwise glue ----------------
__global__ void k_flat(const float* __restrict__ a, const float* __restrict__ b,
                       const float* __restrict__ c, float* __restrict__ flat, int total)
{
    int i = blockIdx.x * blockDim.x + threadIdx.x;
    if (i >= total) return;
    int d = i % 1024, r = i / 1024;
    int ch = r % 3, bb = r / 3;
    const float* src = (ch == 0) ? a : (ch == 1) ? b : c;
    flat[i] = src[bb * 1024 + d];
}

__global__ void k_big(const float* __restrict__ flat, const float* __restrict__ offs,
                      float* __restrict__ big, int total)
{
    int i = blockIdx.x * blockDim.x + threadIdx.x;
    if (i >= total) return;
    const int chunk = CB * 3 * 1024;
    int n = i / chunk, rem = i % chunk;
    big[i] = flat[rem] + offs[n * 3072 + (rem % 3072)];
}

__global__ void k_mean3(const float* __restrict__ x, float* __restrict__ y, int rows)
{
    int i = blockIdx.x * blockDim.x + threadIdx.x;
    if (i >= rows * 1024) return;
    int d = i % 1024, r = i / 1024;
    const float* xr = x + (size_t)r * 3072 + d;
    y[i] = (xr[0] + xr[1024] + xr[2048]) * (1.0f / 3.0f);
}

__global__ void k_build_st(const float* __restrict__ seq, const float* __restrict__ pos,
                           float* __restrict__ st, int t, int total)
{
    int i = blockIdx.x * blockDim.x + threadIdx.x;
    if (i >= total) return;
    int d = i % 1024, r = i / 1024;
    int ti = r % t, bt = r / t;
    st[i] = seq[((size_t)ti * CBT + bt) * 1024 + d] + pos[ti * 1024 + d];
}

__global__ void k_zero(float* __restrict__ p, int n)
{
    int i = blockIdx.x * blockDim.x + threadIdx.x;
    if (i < n) p[i] = 0.f;
}

__global__ void k_accum(const float* __restrict__ tpo, const float* __restrict__ wts,
                        float* __restrict__ out, int e, int total)
{
    // out dims (ND=3, STEPS=5, 3, B=512, D=1024); tpo is (STEPS*Bt, 3*D)
    int i = blockIdx.x * blockDim.x + threadIdx.x;
    if (i >= total) return;
    int d = i % 1024; int r = i / 1024;
    int b = r % CB;   r /= CB;
    int c = r % 3;    r /= 3;
    int s = r % CST;  int n = r / CST;
    int bt = n * CB + b;
    float v = tpo[(((size_t)s * CBT + bt) * 3 + c) * 1024 + d];
    out[i] += wts[b * 4 + e] * v;
}

// ---------------- host orchestration ----------------
static inline int nblk(long n) { return (int)((n + 255) / 256); }

extern "C" void kernel_launch(void* const* d_in, const int* in_sizes, int n_in,
                              void* d_out, int out_size, void* d_ws, size_t ws_size,
                              hipStream_t stream)
{
    const float* iw   = (const float*)d_in[0];   // initial_what (B,D)
    const float* ia   = (const float*)d_in[1];   // initial_action
    const float* ir   = (const float*)d_in[2];   // initial_result
    const float* doff = (const float*)d_in[3];   // dream_offsets (ND,3,D)
    const float* gw1  = (const float*)d_in[4];   // (D,3D)
    const float* gb1  = (const float*)d_in[5];
    const float* glnw = (const float*)d_in[6];
    const float* glnb = (const float*)d_in[7];
    const float* gw2  = (const float*)d_in[8];   // (E,D)
    const float* gb2  = (const float*)d_in[9];
    const float* pos  = (const float*)d_in[10];  // (E,10,D)
    const float* anw  = (const float*)d_in[11];  // (E,L,D)
    const float* anb  = (const float*)d_in[12];
    const float* inw  = (const float*)d_in[13];  // (E,L,3D,D)
    const float* inb  = (const float*)d_in[14];  // (E,L,3D)
    const float* ow   = (const float*)d_in[15];  // (E,L,D,D)
    const float* ob   = (const float*)d_in[16];
    const float* fnw  = (const float*)d_in[17];
    const float* fnb  = (const float*)d_in[18];
    const float* w1   = (const float*)d_in[19];  // (E,L,FF,D)
    const float* b1   = (const float*)d_in[20];
    const float* w2   = (const float*)d_in[21];  // (E,L,D,FF)
    const float* b2   = (const float*)d_in[22];
    const float* tlnw = (const float*)d_in[23];  // (E,D)
    const float* tlnb = (const float*)d_in[24];
    const float* tw1  = (const float*)d_in[25];  // (E,2D,D)
    const float* tb1  = (const float*)d_in[26];
    const float* tw2  = (const float*)d_in[27];  // (E,3D,2D)
    const float* tb2  = (const float*)d_in[28];

    float* ws  = (float*)d_ws;
    float* out = (float*)d_out;

    // workspace layout (floats)
    size_t off = 0;
    size_t FLAT = off; off += (size_t)CB * 3 * CD;        // 1.57M
    size_t G    = off; off += (size_t)CB * CD;            // 0.52M
    size_t LOG  = off; off += (size_t)CB * CE;
    size_t WTS  = off; off += (size_t)CB * CE;
    size_t BIG  = off; off += (size_t)CBT * 3 * CD;       // 4.72M
    size_t SEQ  = off; off += (size_t)6 * CBT * CD;       // 9.44M
    size_t ST   = off; off += (size_t)5 * CBT * CD;       // 7.86M
    size_t XN   = off; off += (size_t)5 * CBT * CD;       // 7.86M
    size_t ATT  = off; off += (size_t)5 * CBT * CD;       // 7.86M
    size_t HBUF = off; off += (size_t)5 * CBT * CFF;      // 31.46M (qkv/ffn/tp-out)
    size_t TPH  = off; off += (size_t)5 * CBT * 2 * CD;   // 15.73M
    (void)ws_size; (void)n_in; (void)in_sizes;

    auto gemm = [&](const float* A, const float* W, const float* bias, const float* Res,
                    float* Out, int M, int N, int K, int act) {
        dim3 g(N / BNT, M / BMT);
        gemm_bf16_wmma<<<g, 128, 0, stream>>>(A, W, bias, Res, Out, M, N, K, act);
    };

    // ---- gating ----
    int flatN = CB * 3 * CD;
    k_flat<<<nblk(flatN), 256, 0, stream>>>(iw, ia, ir, ws + FLAT, flatN);
    gemm(ws + FLAT, gw1, gb1, nullptr, ws + G, CB, CD, 3 * CD, 0);
    k_ln<<<CB, 256, 0, stream>>>(ws + G, ws + G, glnw, glnb, 1024, 0, 1);
    k_logits<<<CB, 256, 0, stream>>>(ws + G, gw2, gb2, ws + LOG);
    k_top2<<<nblk(CB), 256, 0, stream>>>(ws + LOG, ws + WTS);

    // ---- diversified starts ----
    int bigN = CBT * 3 * CD;
    k_big<<<nblk(bigN), 256, 0, stream>>>(ws + FLAT, doff, ws + BIG, bigN);

    // ---- zero output (accumulated over experts) ----
    k_zero<<<nblk(out_size), 256, 0, stream>>>(out, out_size);

    // ---- expert loop ----
    for (int e = 0; e < CE; ++e) {
        const float* pos_e = pos + (size_t)e * 10 * CD;
        // seq[0] = mean over triplet of big
        k_mean3<<<nblk((long)CBT * CD), 256, 0, stream>>>(ws + BIG, ws + SEQ, CBT);

        for (int s = 0; s < CST; ++s) {
            int t = s + 1;
            int rows = CBT * t;
            int stN = rows * CD;
            k_build_st<<<nblk(stN), 256, 0, stream>>>(ws + SEQ, pos_e, ws + ST, t, stN);

            for (int l = 0; l < CL; ++l) {
                int el = e * CL + l;
                // attn block: xn = LN(st); qkv = xn @ in_w^T + in_b
                k_ln<<<rows, 256, 0, stream>>>(ws + ST, ws + XN,
                                               anw + (size_t)el * CD, anb + (size_t)el * CD, 1024, 0, 0);
                gemm(ws + XN, inw + (size_t)el * 3 * CD * CD, inb + (size_t)el * 3 * CD,
                     nullptr, ws + HBUF, rows, 3 * CD, CD, 0);
                k_attn<<<CBT * CH, 128, 0, stream>>>(ws + HBUF, ws + ATT, t);
                // st += attn @ ow^T + ob
                gemm(ws + ATT, ow + (size_t)el * CD * CD, ob + (size_t)el * CD,
                     ws + ST, ws + ST, rows, CD, CD, 0);
                // ffn block
                k_ln<<<rows, 256, 0, stream>>>(ws + ST, ws + XN,
                                               fnw + (size_t)el * CD, fnb + (size_t)el * CD, 1024, 0, 0);
                gemm(ws + XN, w1 + (size_t)el * CFF * CD, b1 + (size_t)el * CFF,
                     nullptr, ws + HBUF, rows, CFF, CD, 1);
                gemm(ws + HBUF, w2 + (size_t)el * CD * CFF, b2 + (size_t)el * CD,
                     ws + ST, ws + ST, rows, CD, CFF, 0);
            }
            // next = mean3(triplet_proj(st[:, -1]))
            k_ln<<<CBT, 256, 0, stream>>>(ws + ST, ws + XN,
                                          tlnw + (size_t)e * CD, tlnb + (size_t)e * CD,
                                          (long)t * 1024, (long)(t - 1) * 1024, 0);
            gemm(ws + XN, tw1 + (size_t)e * 2 * CD * CD, tb1 + (size_t)e * 2 * CD,
                 nullptr, ws + TPH, CBT, 2 * CD, CD, 1);
            gemm(ws + TPH, tw2 + (size_t)e * 3 * CD * 2 * CD, tb2 + (size_t)e * 3 * CD,
                 nullptr, ws + HBUF, CBT, 3 * CD, 2 * CD, 0);
            k_mean3<<<nblk((long)CBT * CD), 256, 0, stream>>>(ws + HBUF,
                                                              ws + SEQ + (size_t)(s + 1) * CBT * CD, CBT);
        }

        // final triplet projection over embs = seq[1..5]  -> (STEPS*Bt, 3D)
        int R = CST * CBT;
        k_ln<<<R, 256, 0, stream>>>(ws + SEQ + (size_t)CBT * CD, ws + ST,
                                    tlnw + (size_t)e * CD, tlnb + (size_t)e * CD, 1024, 0, 0);
        gemm(ws + ST, tw1 + (size_t)e * 2 * CD * CD, tb1 + (size_t)e * 2 * CD,
             nullptr, ws + TPH, R, 2 * CD, CD, 1);
        gemm(ws + TPH, tw2 + (size_t)e * 3 * CD * 2 * CD, tb2 + (size_t)e * 3 * CD,
             nullptr, ws + HBUF, R, 3 * CD, 2 * CD, 0);

        // weighted accumulate into output
        k_accum<<<nblk(out_size), 256, 0, stream>>>(ws + HBUF, ws + WTS, out, e, out_size);
    }
}
